// FoundationalTimeSeriesModelV4_35278861369698
// MI455X (gfx1250) — matrix operand order repaired and checked by reading, not compiled, not run on hardware
//
#include <hip/hip_runtime.h>
#include <hip/hip_bf16.h>

typedef _Float16 f16;
typedef __attribute__((ext_vector_type(16))) _Float16 v16h;
typedef __attribute__((ext_vector_type(8)))  float    v8f;

#define B_   32
#define L_   512
#define S_   32
#define N_   (B_*S_)
#define P_   64
#define C_   32
#define D_   64
#define NH_  4
#define NL_  2
#define E_   16
#define HID_ 256
#define MO_  128
#define PH_  3
#define FH_  3
#define FEAT_ (C_+D_+MO_)
#define EPS_ 1e-5f

// ---------- wave32 reductions ----------
__device__ inline float wsum(float v) {
  #pragma unroll
  for (int m = 16; m > 0; m >>= 1) v += __shfl_xor(v, m, 32);
  return v;
}
__device__ inline float wmaxr(float v) {
  #pragma unroll
  for (int m = 16; m > 0; m >>= 1) v = fmaxf(v, __shfl_xor(v, m, 32));
  return v;
}

// ---------- 1) RevIN + projection + positional encoding -> act0 (n,L,P) f16 ----------
__global__ void revin_proj_kernel(const float* __restrict__ x, const float* __restrict__ mask,
                                  const float* __restrict__ rw, const float* __restrict__ rb,
                                  const float* __restrict__ pw, const float* __restrict__ pb,
                                  f16* __restrict__ act, float* __restrict__ last_val) {
  __shared__ float r1[256], r2[256];
  __shared__ float sh_xn[L_];
  __shared__ float s_mean, s_rstd;
  const int n = blockIdx.x;
  const int b = n / S_, s = n % S_;
  const int tid = threadIdx.x;
  const float m = mask[n];
  float sum = 0.f, sq = 0.f;
  for (int l = tid; l < L_; l += 256) {
    float v = x[((size_t)b*L_ + l)*S_ + s] * m;
    sum += v; sq += v*v;
  }
  r1[tid] = sum; r2[tid] = sq;
  __syncthreads();
  for (int st = 128; st > 0; st >>= 1) {
    if (tid < st) { r1[tid] += r1[tid+st]; r2[tid] += r2[tid+st]; }
    __syncthreads();
  }
  if (tid == 0) {
    float mean = r1[0] / (float)L_;
    float var  = r2[0] / (float)L_ - mean*mean;
    s_mean = mean;
    s_rstd = rsqrtf(var + EPS_);
  }
  __syncthreads();
  const float mean = s_mean, rstd = s_rstd;
  const float rwv = rw[s], rbv = rb[s];
  for (int l = tid; l < L_; l += 256) {
    float v = x[((size_t)b*L_ + l)*S_ + s] * m;
    sh_xn[l] = (v - mean) * rstd * rwv + rbv;
  }
  __syncthreads();
  if (tid == 0) last_val[n] = sh_xn[L_-1];
  const float LOGC = -9.210340371976184f / (float)P_;  // -ln(10000)/P
  for (int idx = tid; idx < L_*P_; idx += 256) {
    int l = idx >> 6, p = idx & 63;
    int j = p >> 1;
    float div = expf((float)(2*j) * LOGC);
    float arg = (float)l * div;
    float pe  = (p & 1) ? cosf(arg) : sinf(arg);
    float v = sh_xn[l]*pw[p] + pb[p] + pe;
    act[((size_t)n*L_ + l)*P_ + p] = (f16)v;
  }
}

// ---------- 2) repack conv weights into WMMA 16-bit A-matrix lane layout ----------
// A (16x32 f16) lane layout used here (contiguous k, as for B): M = lane%16, half = lane/16,
// element e: K = half*16 + e
__global__ void repack_w_kernel(const float* __restrict__ w1, const float* __restrict__ w2,
                                const float* __restrict__ w3, const float* __restrict__ w4,
                                f16* __restrict__ wrep) {
  const int g = blockIdx.x;       // 0..83 tiles
  const int lane = threadIdx.x;   // 0..31
  int ti, mtiles;
  const float* W;
  int lbase;
  if (g < 24)      { ti = g;      mtiles = 4; W = w1; lbase = 0;     }
  else if (g < 48) { ti = g - 24; mtiles = 4; W = w2; lbase = 12288; }
  else if (g < 72) { ti = g - 48; mtiles = 4; W = w3; lbase = 24576; }
  else             { ti = g - 72; mtiles = 2; W = w4; lbase = 36864; }
  const int tap   = ti / (2*mtiles);
  const int rem   = ti % (2*mtiles);
  const int chunk = rem / mtiles;
  const int mt    = rem % mtiles;
  const int o    = mt*16 + (lane & 15);
  const int half = lane >> 4;
  f16* dst = wrep + lbase + ti*512 + lane*16;
  #pragma unroll
  for (int e = 0; e < 16; ++e) {
    int kk = half*16 + e;
    int i  = chunk*32 + kk;
    dst[e] = (f16)W[(o*64 + i)*3 + tap];
  }
}

// ---------- 3) causal dilated conv as WMMA GEMM ----------
// y[o,t] = relu(bias[o] + sum_{tap,i} W[o,i,tap] * xin[i, t-(2-tap)*dil])
// One wave per (n, time-tile): B fragments loaded ONCE, reused across all
// MTILES out-channel tiles -> MTILES*6 v_wmma_f32_16x16x32_f16 per wave.
template <int MTILES, bool F32OUT>
__global__ void conv_wmma_kernel(const f16* __restrict__ ain, void* __restrict__ outp,
                                 const f16* __restrict__ wrep, const float* __restrict__ bias,
                                 int dil) {
  constexpr int COUT = MTILES * 16;
  const int lane  = threadIdx.x;
  const int tt    = blockIdx.x;
  const int n     = blockIdx.z;
  const int tbase = tt * 16;
  const int half  = lane >> 4;
  const int nn    = lane & 15;

  v16h bz;
  #pragma unroll
  for (int i = 0; i < 16; ++i) bz[i] = (f16)0.0f;

  // Load the 6 B fragments (3 taps x 2 K-chunks) once for this time-tile.
  // B layout: K = half*16 + e (contiguous channels), N = lane%16.
  v16h bfrag[6];
  #pragma unroll
  for (int tap = 0; tap < 3; ++tap) {
    const int t = tbase + nn - (2 - tap)*dil;   // causal shift
    #pragma unroll
    for (int c = 0; c < 2; ++c) {
      v16h b = bz;
      if (t >= 0) b = *(const v16h*)(ain + ((size_t)n*L_ + t)*64 + c*32 + half*16);
      bfrag[tap*2 + c] = b;
    }
  }
  // Warm WGP$/L2 for the next time-tile (global_prefetch_b8).
  if (tbase + 16 < L_) {
    __builtin_prefetch(ain + ((size_t)n*L_ + tbase + 16 + nn)*64 + half*16, 0, 1);
  }

  const int tcol = tbase + nn;
  #pragma unroll
  for (int mt = 0; mt < MTILES; ++mt) {
    v8f acc;
    #pragma unroll
    for (int r = 0; r < 8; ++r) acc[r] = 0.0f;
    #pragma unroll
    for (int k = 0; k < 6; ++k) {
      v16h a = *(const v16h*)(wrep + (size_t)(k*MTILES + mt)*512 + lane*16);
      acc = __builtin_amdgcn_wmma_f32_16x16x32_f16(false, a, false, bfrag[k],
                                                   (short)0, acc, false, false);
    }
    #pragma unroll
    for (int r = 0; r < 8; ++r) {
      const int o = mt*16 + half*8 + r;          // D layout: M = r + 8*half
      float v = fmaxf(acc[r] + bias[o], 0.0f);
      if (F32OUT) ((float*)outp)[((size_t)n*L_ + tcol)*COUT + o] = v;
      else        ((f16*)  outp)[((size_t)n*L_ + tcol)*COUT + o] = (f16)v;
    }
  }
}

// ---------- 4) LayerNorm over C=32 + mask + mean-pool over L + last-step feats ----------
__global__ void lnpool_kernel(const float* __restrict__ h, const float* __restrict__ mask,
                              const float* __restrict__ g, const float* __restrict__ bta,
                              float* __restrict__ pooled, float* __restrict__ flast) {
  const int n = blockIdx.x;
  const int c = threadIdx.x;        // channel = lane (wave32 fits C=32 exactly)
  const float m  = mask[n];
  const float gc = g[c], bc = bta[c];
  float acc = 0.0f, lastv = 0.0f;
  for (int l = 0; l < L_; ++l) {
    float v   = h[((size_t)n*L_ + l)*C_ + c];
    float sum = wsum(v);
    float sq  = wsum(v*v);
    float mean = sum * (1.0f/C_);
    float var  = sq  * (1.0f/C_) - mean*mean;
    float f = ((v - mean) * rsqrtf(var + EPS_) * gc + bc) * m;
    acc += f;
    if (l == L_-1) lastv = f;
  }
  pooled[n*C_ + c] = acc * (1.0f/L_);
  flast [n*C_ + c] = lastv;
}

// ---------- 5) pooled @ p2t_w.T + b + pos_inter -> t ----------
__global__ void p2t_kernel(const float* __restrict__ pooled, const float* __restrict__ w,
                           const float* __restrict__ bias, const float* __restrict__ pos,
                           float* __restrict__ t) {
  int idx = blockIdx.x*blockDim.x + threadIdx.x;
  if (idx >= N_*D_) return;
  int n = idx / D_, d = idx % D_;
  int s = n % S_;
  float acc = bias[d] + pos[s*D_ + d];
  #pragma unroll 4
  for (int c = 0; c < C_; ++c) acc += pooled[n*C_ + c] * w[d*C_ + c];
  t[idx] = acc;
}

// ---------- 6) whole transformer stack per batch element, in LDS ----------
__global__ void transformer_kernel(const float* __restrict__ tin, const float* __restrict__ mask,
    const float* __restrict__ ln1g, const float* __restrict__ ln1b,
    const float* __restrict__ qkvw, const float* __restrict__ qkvb,
    const float* __restrict__ aow,  const float* __restrict__ aob,
    const float* __restrict__ ln2g, const float* __restrict__ ln2b,
    const float* __restrict__ f1w,  const float* __restrict__ f1b,
    const float* __restrict__ f2w,  const float* __restrict__ f2b,
    const float* __restrict__ trg,  const float* __restrict__ trb,
    float* __restrict__ ctx, float* __restrict__ gin) {
  __shared__ float ts[S_*D_];
  __shared__ float xs[S_*D_];
  __shared__ float qk[S_*3*D_];
  __shared__ float os[S_*D_];
  __shared__ float sc[S_*S_];
  __shared__ float msk[S_];
  const int b = blockIdx.x;
  const int tid = threadIdx.x;
  const int lane = tid & 31;
  const int wv = tid >> 5;
  for (int i = tid; i < S_*D_; i += 256) ts[i] = tin[(size_t)b*S_*D_ + i];
  if (tid < S_) msk[tid] = mask[b*S_ + tid];
  __syncthreads();

  for (int layer = 0; layer < NL_; ++layer) {
    // LN1 -> xs (one wave per row, lane j handles dims j, j+32)
    {
      const float* g1 = ln1g + layer*D_; const float* b1 = ln1b + layer*D_;
      for (int row = wv; row < S_; row += 8) {
        float v0 = ts[row*D_ + lane], v1 = ts[row*D_ + lane + 32];
        float sum = wsum(v0 + v1), sq = wsum(v0*v0 + v1*v1);
        float mean = sum*(1.0f/D_), var = sq*(1.0f/D_) - mean*mean;
        float rs = rsqrtf(var + EPS_);
        xs[row*D_ + lane]      = (v0-mean)*rs*g1[lane]    + b1[lane];
        xs[row*D_ + lane + 32] = (v1-mean)*rs*g1[lane+32] + b1[lane+32];
      }
    }
    __syncthreads();
    // QKV projection
    {
      const float* wq = qkvw + (size_t)layer*3*D_*D_;
      const float* bq = qkvb + layer*3*D_;
      for (int idx = tid; idx < S_*3*D_; idx += 256) {
        int r = idx/(3*D_), c = idx%(3*D_);
        float acc = bq[c];
        const float* xr = xs + r*D_;
        const float* wc = wq + c*D_;
        #pragma unroll 4
        for (int j = 0; j < D_; ++j) acc += xr[j]*wc[j];
        qk[idx] = acc;
      }
    }
    __syncthreads();
    for (int i = tid; i < S_*D_; i += 256) os[i] = 0.0f;
    __syncthreads();
    // attention per head
    for (int h = 0; h < NH_; ++h) {
      const int hb = h*16;
      for (int idx = tid; idx < S_*S_; idx += 256) {
        int qr = idx/S_, kr = idx%S_;
        float acc = 0.0f;
        #pragma unroll
        for (int d = 0; d < 16; ++d) acc += qk[qr*192 + hb + d] * qk[kr*192 + 64 + hb + d];
        acc *= 0.25f;                       // 1/sqrt(16)
        if (msk[kr] == 0.0f) acc = -1e9f;
        sc[idx] = acc;
      }
      __syncthreads();
      for (int row = wv; row < S_; row += 8) {
        float v = sc[row*S_ + lane];
        float mx = wmaxr(v);
        float e = expf(v - mx);
        float se = wsum(e);
        sc[row*S_ + lane] = e / se;
      }
      __syncthreads();
      for (int idx = tid; idx < S_*16; idx += 256) {
        int qr = idx/16, d = idx%16;
        float acc = 0.0f;
        for (int kr = 0; kr < S_; ++kr) acc += sc[qr*S_ + kr] * qk[kr*192 + 128 + hb + d];
        os[qr*D_ + hb + d] = acc;
      }
      __syncthreads();
    }
    // out-proj + residual
    {
      const float* wo = aow + (size_t)layer*D_*D_;
      const float* bo = aob + layer*D_;
      for (int idx = tid; idx < S_*D_; idx += 256) {
        int r = idx/D_, c = idx%D_;
        float acc = bo[c];
        #pragma unroll 4
        for (int j = 0; j < D_; ++j) acc += os[r*D_ + j]*wo[c*D_ + j];
        ts[idx] += acc;
      }
    }
    __syncthreads();
    // LN2 -> xs
    {
      const float* g2 = ln2g + layer*D_; const float* b2 = ln2b + layer*D_;
      for (int row = wv; row < S_; row += 8) {
        float v0 = ts[row*D_ + lane], v1 = ts[row*D_ + lane + 32];
        float sum = wsum(v0 + v1), sq = wsum(v0*v0 + v1*v1);
        float mean = sum*(1.0f/D_), var = sq*(1.0f/D_) - mean*mean;
        float rs = rsqrtf(var + EPS_);
        xs[row*D_ + lane]      = (v0-mean)*rs*g2[lane]    + b2[lane];
        xs[row*D_ + lane + 32] = (v1-mean)*rs*g2[lane+32] + b2[lane+32];
      }
    }
    __syncthreads();
    // FF1 (relu) -> reuse qk as [32][128]
    {
      const float* w1 = f1w + (size_t)layer*2*D_*D_;
      const float* bb = f1b + layer*2*D_;
      for (int idx = tid; idx < S_*2*D_; idx += 256) {
        int r = idx/(2*D_), c = idx%(2*D_);
        float acc = bb[c];
        #pragma unroll 4
        for (int j = 0; j < D_; ++j) acc += xs[r*D_ + j]*w1[c*D_ + j];
        qk[r*2*D_ + c] = fmaxf(acc, 0.0f);
      }
    }
    __syncthreads();
    // FF2 + residual
    {
      const float* w2 = f2w + (size_t)layer*D_*2*D_;
      const float* bb = f2b + layer*D_;
      for (int idx = tid; idx < S_*D_; idx += 256) {
        int r = idx/D_, c = idx%D_;
        float acc = bb[c];
        #pragma unroll 4
        for (int j = 0; j < 2*D_; ++j) acc += qk[r*2*D_ + j]*w2[c*2*D_ + j];
        ts[idx] += acc;
      }
    }
    __syncthreads();
  }
  // final LN * mask -> ctx (global + xs)
  for (int row = wv; row < S_; row += 8) {
    float v0 = ts[row*D_ + lane], v1 = ts[row*D_ + lane + 32];
    float sum = wsum(v0 + v1), sq = wsum(v0*v0 + v1*v1);
    float mean = sum*(1.0f/D_), var = sq*(1.0f/D_) - mean*mean;
    float rs = rsqrtf(var + EPS_);
    float c0 = ((v0-mean)*rs*trg[lane]    + trb[lane])    * msk[row];
    float c1 = ((v1-mean)*rs*trg[lane+32] + trb[lane+32]) * msk[row];
    xs[row*D_ + lane] = c0;
    xs[row*D_ + lane + 32] = c1;
    ctx[(size_t)b*S_*D_ + row*D_ + lane]      = c0;
    ctx[(size_t)b*S_*D_ + row*D_ + lane + 32] = c1;
  }
  __syncthreads();
  // gin = sum_s(ctx*mask)/clip(sum mask,1)
  if (tid < D_) {
    float denom = 0.0f;
    for (int s = 0; s < S_; ++s) denom += msk[s];
    denom = fmaxf(denom, 1.0f);
    float acc = 0.0f;
    for (int s = 0; s < S_; ++s) acc += xs[s*D_ + tid] * msk[s];
    gin[b*D_ + tid] = acc / denom;
  }
}

// ---------- 7) expert MLPs: eo[b,e,:] ----------
__global__ void experts_kernel(const float* __restrict__ gin,
                               const float* __restrict__ w1, const float* __restrict__ b1,
                               const float* __restrict__ w2, const float* __restrict__ b2,
                               float* __restrict__ eo) {
  __shared__ float h1[HID_];
  __shared__ float gsh[D_];
  const int e = blockIdx.x, b = blockIdx.y;
  const int tid = threadIdx.x;   // 256 == HID_
  if (tid < D_) gsh[tid] = gin[b*D_ + tid];
  __syncthreads();
  {
    float acc = b1[e*HID_ + tid];
    const float* wr = w1 + ((size_t)e*HID_ + tid)*D_;
    #pragma unroll 4
    for (int d = 0; d < D_; ++d) acc += gsh[d]*wr[d];
    h1[tid] = fmaxf(acc, 0.0f);
  }
  __syncthreads();
  if (tid < MO_) {
    float acc = b2[e*MO_ + tid];
    const float* wr = w2 + ((size_t)e*MO_ + tid)*HID_;
    #pragma unroll 4
    for (int h = 0; h < HID_; ++h) acc += h1[h]*wr[h];
    eo[((size_t)b*E_ + e)*MO_ + tid] = acc;
  }
}

// ---------- 8) three top-2 MoE gates -> mf, ml, mr ----------
__global__ void moe_kernel(const float* __restrict__ gin, const float* __restrict__ eo,
                           const float* __restrict__ gfw, const float* __restrict__ gfb,
                           const float* __restrict__ glw, const float* __restrict__ glb,
                           const float* __restrict__ grw, const float* __restrict__ grb,
                           float* __restrict__ m3) {
  __shared__ float lg[E_];
  __shared__ int si0, si1;
  __shared__ float sg0, sg1;
  const int b = blockIdx.x;
  const int tid = threadIdx.x;   // 128
  const float* gws[3] = {gfw, glw, grw};
  const float* gbs[3] = {gfb, glb, grb};
  for (int g = 0; g < 3; ++g) {
    if (tid < E_) {
      float acc = gbs[g][tid];
      for (int d = 0; d < D_; ++d) acc += gin[b*D_ + d]*gws[g][tid*D_ + d];
      lg[tid] = acc;
    }
    __syncthreads();
    if (tid == 0) {
      int i0 = 0; float v0 = lg[0];
      for (int e = 1; e < E_; ++e) if (lg[e] > v0) { v0 = lg[e]; i0 = e; }
      int i1 = 0; float v1 = -1e30f;
      for (int e = 0; e < E_; ++e) if (e != i0 && lg[e] > v1) { v1 = lg[e]; i1 = e; }
      float e1 = expf(v1 - v0);           // softmax over [v0, v1], max = v0
      si0 = i0; si1 = i1;
      sg0 = 1.0f/(1.0f + e1);
      sg1 = e1 /(1.0f + e1);
    }
    __syncthreads();
    m3[(size_t)g*B_*MO_ + b*MO_ + tid] =
        sg0*eo[((size_t)b*E_ + si0)*MO_ + tid] + sg1*eo[((size_t)b*E_ + si1)*MO_ + tid];
    __syncthreads();
  }
}

// ---------- 9) output heads -> d_out = [pred | fail | rca] ----------
__global__ void heads_kernel(const float* __restrict__ flast, const float* __restrict__ ctx,
                             const float* __restrict__ mf, const float* __restrict__ ml,
                             const float* __restrict__ mr, const float* __restrict__ lastv,
                             const float* __restrict__ pw, const float* __restrict__ pb,
                             const float* __restrict__ fw, const float* __restrict__ fb,
                             const float* __restrict__ rw, const float* __restrict__ rb,
                             float* __restrict__ out) {
  const int idx = blockIdx.x*blockDim.x + threadIdx.x;
  const int TOT = N_*PH_ + B_*FH_ + N_;
  if (idx >= TOT) return;
  if (idx < N_*PH_) {
    int n = idx / PH_, ph = idx % PH_;
    int b = n / S_;
    const float* w = pw + ph*FEAT_;
    float acc = pb[ph];
    for (int f = 0; f < C_;  ++f) acc += flast[n*C_ + f] * w[f];
    for (int f = 0; f < D_;  ++f) acc += ctx[n*D_ + f]   * w[C_ + f];
    for (int f = 0; f < MO_; ++f) acc += mf[b*MO_ + f]   * w[C_ + D_ + f];
    out[idx] = lastv[n] + acc;
  } else if (idx < N_*PH_ + B_*FH_) {
    int j = idx - N_*PH_;
    int b = j / FH_, fh = j % FH_;
    float acc = fb[fh];
    for (int o = 0; o < MO_; ++o) acc += ml[b*MO_ + o] * fw[fh*MO_ + o];
    out[idx] = acc;
  } else {
    int n = idx - N_*PH_ - B_*FH_;
    int b = n / S_;
    float acc = rb[0];
    for (int f = 0; f < C_;  ++f) acc += flast[n*C_ + f] * rw[f];
    for (int f = 0; f < D_;  ++f) acc += ctx[n*D_ + f]   * rw[C_ + f];
    for (int f = 0; f < MO_; ++f) acc += mr[b*MO_ + f]   * rw[C_ + D_ + f];
    out[idx] = acc;
  }
}

extern "C" void kernel_launch(void* const* d_in, const int* in_sizes, int n_in,
                              void* d_out, int out_size, void* d_ws, size_t ws_size,
                              hipStream_t stream) {
  (void)in_sizes; (void)n_in; (void)out_size; (void)ws_size;
  const float* x       = (const float*)d_in[0];
  const float* mask    = (const float*)d_in[1];
  const float* revin_w = (const float*)d_in[2];
  const float* revin_b = (const float*)d_in[3];
  const float* proj_w  = (const float*)d_in[4];
  const float* proj_b  = (const float*)d_in[5];
  const float* cw1 = (const float*)d_in[6];  const float* cb1 = (const float*)d_in[7];
  const float* cw2 = (const float*)d_in[8];  const float* cb2 = (const float*)d_in[9];
  const float* cw3 = (const float*)d_in[10]; const float* cb3 = (const float*)d_in[11];
  const float* cw4 = (const float*)d_in[12]; const float* cb4 = (const float*)d_in[13];
  const float* fn_g = (const float*)d_in[14]; const float* fn_b = (const float*)d_in[15];
  const float* p2t_w = (const float*)d_in[16]; const float* p2t_b = (const float*)d_in[17];
  const float* pos_inter = (const float*)d_in[18];
  const float* ln1_g = (const float*)d_in[19]; const float* ln1_b = (const float*)d_in[20];
  const float* qkv_w = (const float*)d_in[21]; const float* qkv_b = (const float*)d_in[22];
  const float* attn_ow = (const float*)d_in[23]; const float* attn_ob = (const float*)d_in[24];
  const float* ln2_g = (const float*)d_in[25]; const float* ln2_b = (const float*)d_in[26];
  const float* ff1_w = (const float*)d_in[27]; const float* ff1_b = (const float*)d_in[28];
  const float* ff2_w = (const float*)d_in[29]; const float* ff2_b = (const float*)d_in[30];
  const float* tr_g = (const float*)d_in[31]; const float* tr_b = (const float*)d_in[32];
  const float* exp_w1 = (const float*)d_in[33]; const float* exp_b1 = (const float*)d_in[34];
  const float* exp_w2 = (const float*)d_in[35]; const float* exp_b2 = (const float*)d_in[36];
  const float* gf_w = (const float*)d_in[37]; const float* gf_b = (const float*)d_in[38];
  const float* gl_w = (const float*)d_in[39]; const float* gl_b = (const float*)d_in[40];
  const float* gr_w = (const float*)d_in[41]; const float* gr_b = (const float*)d_in[42];
  const float* pred_w = (const float*)d_in[43]; const float* pred_b = (const float*)d_in[44];
  const float* fail_w = (const float*)d_in[45]; const float* fail_b = (const float*)d_in[46];
  const float* rca_w = (const float*)d_in[47]; const float* rca_b = (const float*)d_in[48];

  char* ws = (char*)d_ws;
  size_t off = 0;
  auto alloc = [&](size_t bytes) -> void* {
    void* p = ws + off;
    off += (bytes + 255) & ~(size_t)255;
    return p;
  };
  f16*   buf0   = (f16*)  alloc((size_t)N_*L_*64*sizeof(f16));   // 67 MB (also conv4 f32 out)
  f16*   buf1   = (f16*)  alloc((size_t)N_*L_*64*sizeof(f16));   // 67 MB
  f16*   wrep   = (f16*)  alloc((size_t)43008*sizeof(f16));      // repacked conv weights
  float* lastv  = (float*)alloc((size_t)N_*sizeof(float));
  float* pooled = (float*)alloc((size_t)N_*C_*sizeof(float));
  float* flast  = (float*)alloc((size_t)N_*C_*sizeof(float));
  float* tbuf   = (float*)alloc((size_t)N_*D_*sizeof(float));
  float* ctx    = (float*)alloc((size_t)N_*D_*sizeof(float));
  float* gin    = (float*)alloc((size_t)B_*D_*sizeof(float));
  float* eo     = (float*)alloc((size_t)B_*E_*MO_*sizeof(float));
  float* m3     = (float*)alloc((size_t)3*B_*MO_*sizeof(float));
  float* mf = m3;
  float* ml = m3 + (size_t)B_*MO_;
  float* mr = m3 + (size_t)2*B_*MO_;

  // 1) RevIN + projection + PE
  revin_proj_kernel<<<N_, 256, 0, stream>>>(x, mask, revin_w, revin_b, proj_w, proj_b,
                                            buf0, lastv);
  // 2) repack all conv weights to WMMA A layout
  repack_w_kernel<<<84, 32, 0, stream>>>(cw1, cw2, cw3, cw4, wrep);
  // 3) conv stack (WMMA): dil 1,2,4,8; one wave per (n, time-tile), B reused
  //    across out-channel tiles; last layer stores f32 x 32ch
  conv_wmma_kernel<4,false><<<dim3(L_/16, 1, N_), 32, 0, stream>>>(buf0, (void*)buf1, wrep,         cb1, 1);
  conv_wmma_kernel<4,false><<<dim3(L_/16, 1, N_), 32, 0, stream>>>(buf1, (void*)buf0, wrep + 12288, cb2, 2);
  conv_wmma_kernel<4,false><<<dim3(L_/16, 1, N_), 32, 0, stream>>>(buf0, (void*)buf1, wrep + 24576, cb3, 4);
  conv_wmma_kernel<2,true ><<<dim3(L_/16, 1, N_), 32, 0, stream>>>(buf1, (void*)buf0, wrep + 36864, cb4, 8);
  // 4) LN over C + mask + pool + last feats
  lnpool_kernel<<<N_, 32, 0, stream>>>((const float*)buf0, mask, fn_g, fn_b, pooled, flast);
  // 5) pooled -> tokens
  p2t_kernel<<<(N_*D_ + 255)/256, 256, 0, stream>>>(pooled, p2t_w, p2t_b, pos_inter, tbuf);
  // 6) transformer + final LN + gin
  transformer_kernel<<<B_, 256, 0, stream>>>(tbuf, mask, ln1_g, ln1_b, qkv_w, qkv_b,
                                             attn_ow, attn_ob, ln2_g, ln2_b,
                                             ff1_w, ff1_b, ff2_w, ff2_b, tr_g, tr_b,
                                             ctx, gin);
  // 7) experts
  experts_kernel<<<dim3(E_, B_), HID_, 0, stream>>>(gin, exp_w1, exp_b1, exp_w2, exp_b2, eo);
  // 8) MoE gating (3 gates)
  moe_kernel<<<B_, MO_, 0, stream>>>(gin, eo, gf_w, gf_b, gl_w, gl_b, gr_w, gr_b, m3);
  // 9) heads
  const int TOT = N_*PH_ + B_*FH_ + N_;
  heads_kernel<<<(TOT + 255)/256, 256, 0, stream>>>(flast, ctx, mf, ml, mr, lastv,
                                                    pred_w, pred_b, fail_w, fail_b,
                                                    rca_w, rca_b, (float*)d_out);
}
